// GRUDecoder_1486058684772
// MI455X (gfx1250) — matrix operand order repaired and checked by reading, not compiled, not run on hardware
//
#include <hip/hip_runtime.h>

#define BATCH 512
#define DLAT  292
#define DCHAR 35
#define HDIM  501
#define TSTEPS 120
#define N3H   1503          // 3*H
#define NPAD  1504          // N3H padded to multiple of 16
#define KPAD  512           // H padded to multiple of 32
#define NTILES (NPAD / 16)  // 94

typedef __attribute__((ext_vector_type(16))) _Float16 v16h;
typedef __attribute__((ext_vector_type(8)))  _Float16 v8h;
typedef __attribute__((ext_vector_type(8)))  float    v8f;

// ---------------------------------------------------------------------------
// A-fragment loader: 16-bit A matrix 16x32 (MxK) layout, contiguous k order used here (B matches):
//   lanes 0-15  hold M=lane,    K = k0+0..15
//   lanes 16-31 hold M=lane-16, K = k0+16..31
// ---------------------------------------------------------------------------
__device__ __forceinline__ v16h load_a_frag(const _Float16* __restrict__ row,
                                            int k0, int a_ko) {
  v8h lo = *(const v8h*)(row + k0 + a_ko);
  v8h hi = *(const v8h*)(row + k0 + 8 + a_ko);
  return __builtin_shufflevector(lo, hi, 0, 1, 2, 3, 4, 5, 6, 7,
                                         8, 9, 10, 11, 12, 13, 14, 15);
}

// ---------------------------------------------------------------------------
// Fused recurrent GEMM: gh = A_hh @ Whh^T + b_hh  (and, if DUAL,
//                       gi = A_ih @ Wih^T + b_ih) for one timestep/layer.
// A buffers: f16 [BATCH x KPAD] row-major (h states, pad cols zero).
// W buffers: f16 [NPAD x KPAD] row-major (W[n,k], zero padded) -> B fragment
// contiguous k order, as in A: lane<16 holds (N=lane, K=k0..k0+15), lane>=16 (N=lane-16, K=k0+16..31)
// i.e. 16 contiguous halfs of row n -> one aligned 32B load per lane.
// ---------------------------------------------------------------------------
template <int DUAL>
__global__ __launch_bounds__(256) void gemm_step(
    const _Float16* __restrict__ Ahh, const _Float16* __restrict__ Aih,
    const _Float16* __restrict__ Whh, const _Float16* __restrict__ Wih,
    const float* __restrict__ bhh, const float* __restrict__ bih,
    float* __restrict__ gh, float* __restrict__ gi) {
  const int lane = threadIdx.x & 31;
  const int wv   = threadIdx.x >> 5;
  const int m0   = (blockIdx.y * 8 + wv) * 16;
  const int n0   = blockIdx.x * 16;

  const int am   = m0 + (lane & 15);
  const int a_ko = (lane < 16) ? 0 : 16;
  const int bn   = n0 + (lane & 15);
  const int b_ko = (lane < 16) ? 0 : 16;

  const _Float16* arow1 = Ahh + (size_t)am * KPAD;
  const _Float16* brow1 = Whh + (size_t)bn * KPAD;
  const _Float16* arow2 = DUAL ? (Aih + (size_t)am * KPAD) : nullptr;
  const _Float16* brow2 = DUAL ? (Wih + (size_t)bn * KPAD) : nullptr;

  v8f acc1 = {};
  v8f acc2 = {};

  for (int k0 = 0; k0 < KPAD; k0 += 32) {
    v16h a1 = load_a_frag(arow1, k0, a_ko);
    v16h b1 = *(const v16h*)(brow1 + k0 + b_ko);
    acc1 = __builtin_amdgcn_wmma_f32_16x16x32_f16(false, a1, false, b1,
                                                  (short)0, acc1, false, false);
    if (DUAL) {
      v16h a2 = load_a_frag(arow2, k0, a_ko);
      v16h b2 = *(const v16h*)(brow2 + k0 + b_ko);
      acc2 = __builtin_amdgcn_wmma_f32_16x16x32_f16(false, a2, false, b2,
                                                    (short)0, acc2, false, false);
    }
  }

  // C/D layout: VGPR r, lane<16 -> (M=m0+r,   N=n0+lane)
  //                     lane>=16 -> (M=m0+8+r, N=n0+lane-16)
  const float bias1 = (bn < N3H) ? bhh[bn] : 0.0f;
  const int   mrow  = m0 + ((lane < 16) ? 0 : 8);
  float* ghp = gh + (size_t)mrow * NPAD + bn;
#pragma unroll
  for (int r = 0; r < 8; r++) ghp[(size_t)r * NPAD] = acc1[r] + bias1;
  if (DUAL) {
    const float bias2 = (bn < N3H) ? bih[bn] : 0.0f;
    float* gip = gi + (size_t)mrow * NPAD + bn;
#pragma unroll
    for (int r = 0; r < 8; r++) gip[(size_t)r * NPAD] = acc2[r] + bias2;
  }
}

// ---------------------------------------------------------------------------
// GRU cell pointwise update; writes h as f32 and f16 (WMMA A operand copy).
// ---------------------------------------------------------------------------
__global__ __launch_bounds__(256) void gru_cell(
    const float* __restrict__ gi, int gi_stride, const float* __restrict__ gh,
    float* __restrict__ hf, _Float16* __restrict__ hh) {
  int idx = blockIdx.x * blockDim.x + threadIdx.x;
  if (idx >= BATCH * HDIM) return;
  int b = idx / HDIM;
  int j = idx - b * HDIM;
  const float* gib = gi + (size_t)b * gi_stride;
  const float* ghb = gh + (size_t)b * NPAD;
  float r    = 1.0f / (1.0f + expf(-(gib[j] + ghb[j])));
  float zg   = 1.0f / (1.0f + expf(-(gib[HDIM + j] + ghb[HDIM + j])));
  float n    = tanhf(gib[2 * HDIM + j] + r * ghb[2 * HDIM + j]);
  float hold = hf[(size_t)b * KPAD + j];
  float hn   = (1.0f - zg) * n + zg * hold;
  hf[(size_t)b * KPAD + j] = hn;
  hh[(size_t)b * KPAD + j] = (_Float16)hn;
}

// ---------------------------------------------------------------------------
// One-time: x = selu(z @ fc1_w^T + fc1_b)
// ---------------------------------------------------------------------------
__global__ __launch_bounds__(256) void fc1_selu(const float* __restrict__ z,
                                                const float* __restrict__ w,
                                                const float* __restrict__ bias,
                                                float* __restrict__ x) {
  int idx = blockIdx.x * blockDim.x + threadIdx.x;
  if (idx >= BATCH * DLAT) return;
  int b = idx / DLAT;
  int i = idx - b * DLAT;
  float s = bias[i];
  const float* zb = z + (size_t)b * DLAT;
  const float* wi = w + (size_t)i * DLAT;
  for (int k = 0; k < DLAT; k++) s += zb[k] * wi[k];
  const float alpha = 1.6732632423543772f, scale = 1.0507009873554805f;
  x[idx] = scale * (s > 0.0f ? s : alpha * (expf(s) - 1.0f));
}

// ---------------------------------------------------------------------------
// One-time: gi0 = x @ w_ih0^T + b_ih0 (kept fp32, reused all 120 steps)
// ---------------------------------------------------------------------------
__global__ __launch_bounds__(256) void gi0_kernel(
    const float* __restrict__ x, const float* __restrict__ w,
    const float* __restrict__ bias, float* __restrict__ gi0) {
  int idx = blockIdx.x * blockDim.x + threadIdx.x;
  if (idx >= BATCH * N3H) return;
  int b = idx / N3H;
  int j = idx - b * N3H;
  float s = bias[j];
  const float* xb = x + (size_t)b * DLAT;
  const float* wj = w + (size_t)j * DLAT;
  for (int k = 0; k < DLAT; k++) s += xb[k] * wj[k];
  gi0[idx] = s;
}

// ---------------------------------------------------------------------------
// Convert 5 fp32 weight matrices [N3H x HDIM] -> f16 [NPAD x KPAD], zero pad.
// ---------------------------------------------------------------------------
__global__ __launch_bounds__(256) void convert_w(
    const float* __restrict__ s0, const float* __restrict__ s1,
    const float* __restrict__ s2, const float* __restrict__ s3,
    const float* __restrict__ s4, _Float16* __restrict__ d0,
    _Float16* __restrict__ d1, _Float16* __restrict__ d2,
    _Float16* __restrict__ d3, _Float16* __restrict__ d4) {
  const float* s;
  _Float16* d;
  switch (blockIdx.z) {
    case 0: s = s0; d = d0; break;
    case 1: s = s1; d = d1; break;
    case 2: s = s2; d = d2; break;
    case 3: s = s3; d = d3; break;
    default: s = s4; d = d4; break;
  }
  int idx = blockIdx.x * blockDim.x + threadIdx.x;
  if (idx >= NPAD * KPAD) return;
  int n = idx / KPAD;
  int k = idx - n * KPAD;
  float v = (n < N3H && k < HDIM) ? s[(size_t)n * HDIM + k] : 0.0f;
  d[idx] = (_Float16)v;
}

__global__ __launch_bounds__(256) void zero_f32(float* __restrict__ p, int n) {
  int i = blockIdx.x * blockDim.x + threadIdx.x;
  if (i < n) p[i] = 0.0f;
}

// ---------------------------------------------------------------------------
// Per-step: out[:, t, :] = softmax(h2 @ fc2_w^T + fc2_b). One wave per batch row.
// ---------------------------------------------------------------------------
__global__ __launch_bounds__(256) void fc2_softmax(
    const float* __restrict__ h2, const float* __restrict__ w,
    const float* __restrict__ bias, float* __restrict__ out, int t) {
  const int lane = threadIdx.x & 31;
  const int wv   = threadIdx.x >> 5;
  const int b    = blockIdx.x * 8 + wv;
  const float* hb = h2 + (size_t)b * KPAD;
  float logit[DCHAR];
#pragma unroll
  for (int o = 0; o < DCHAR; o++) {
    float s = 0.0f;
    for (int k = lane; k < HDIM; k += 32) s += hb[k] * w[(size_t)o * HDIM + k];
#pragma unroll
    for (int d = 16; d > 0; d >>= 1) s += __shfl_xor(s, d, 32);
    logit[o] = s + bias[o];
  }
  float mx = logit[0];
#pragma unroll
  for (int o = 1; o < DCHAR; o++) mx = fmaxf(mx, logit[o]);
  float sum = 0.0f;
#pragma unroll
  for (int o = 0; o < DCHAR; o++) {
    logit[o] = expf(logit[o] - mx);
    sum += logit[o];
  }
  float inv = 1.0f / sum;
  if (lane == 0) {
    float* ob = out + ((size_t)b * TSTEPS + t) * DCHAR;
#pragma unroll
    for (int o = 0; o < DCHAR; o++) ob[o] = logit[o] * inv;
  }
}

// ---------------------------------------------------------------------------
extern "C" void kernel_launch(void* const* d_in, const int* in_sizes, int n_in,
                              void* d_out, int out_size, void* d_ws,
                              size_t ws_size, hipStream_t stream) {
  const float* z      = (const float*)d_in[0];
  const float* fc1_w  = (const float*)d_in[1];
  const float* fc1_b  = (const float*)d_in[2];
  const float* w_ih0  = (const float*)d_in[3];
  const float* w_hh0  = (const float*)d_in[4];
  const float* b_ih0  = (const float*)d_in[5];
  const float* b_hh0  = (const float*)d_in[6];
  const float* w_ih1  = (const float*)d_in[7];
  const float* w_hh1  = (const float*)d_in[8];
  const float* b_ih1  = (const float*)d_in[9];
  const float* b_hh1  = (const float*)d_in[10];
  const float* w_ih2  = (const float*)d_in[11];
  const float* w_hh2  = (const float*)d_in[12];
  const float* b_ih2  = (const float*)d_in[13];
  const float* b_hh2  = (const float*)d_in[14];
  const float* fc2_w  = (const float*)d_in[15];
  const float* fc2_b  = (const float*)d_in[16];
  float* out = (float*)d_out;

  // Workspace carve-up (all chunks 256B aligned; ~22.3 MB total)
  char*  ws  = (char*)d_ws;
  size_t off = 0;
  auto take = [&](size_t bytes) -> char* {
    char* p = ws + off;
    off = (off + bytes + 255) & ~(size_t)255;
    return p;
  };
  _Float16* wh0 = (_Float16*)take((size_t)NPAD * KPAD * 2);
  _Float16* wi1 = (_Float16*)take((size_t)NPAD * KPAD * 2);
  _Float16* wh1 = (_Float16*)take((size_t)NPAD * KPAD * 2);
  _Float16* wi2 = (_Float16*)take((size_t)NPAD * KPAD * 2);
  _Float16* wh2 = (_Float16*)take((size_t)NPAD * KPAD * 2);
  float*    h0f = (float*)take((size_t)BATCH * KPAD * 4);
  float*    h1f = (float*)take((size_t)BATCH * KPAD * 4);
  float*    h2f = (float*)take((size_t)BATCH * KPAD * 4);
  _Float16* h0h = (_Float16*)take((size_t)BATCH * KPAD * 2);
  _Float16* h1h = (_Float16*)take((size_t)BATCH * KPAD * 2);
  _Float16* h2h = (_Float16*)take((size_t)BATCH * KPAD * 2);
  float*    xb  = (float*)take((size_t)BATCH * DLAT * 4);
  float*    gi0 = (float*)take((size_t)BATCH * N3H * 4);
  float*    ghs = (float*)take((size_t)BATCH * NPAD * 4);
  float*    gis = (float*)take((size_t)BATCH * NPAD * 4);

  // 1) Convert recurrent weights to f16 [NPAD x KPAD]
  dim3 cg((NPAD * KPAD + 255) / 256, 1, 5);
  convert_w<<<cg, 256, 0, stream>>>(w_hh0, w_ih1, w_hh1, w_ih2, w_hh2,
                                    wh0, wi1, wh1, wi2, wh2);

  // 2) Zero h state region (h0f..h2h are contiguous: 3*1MB f32 + 3*0.5MB f16)
  int zcount = 3 * BATCH * KPAD + (3 * BATCH * KPAD) / 2;  // in floats
  zero_f32<<<(zcount + 255) / 256, 256, 0, stream>>>(h0f, zcount);

  // 3) x = selu(fc1), gi0 = x @ w_ih0^T + b_ih0 (one-time)
  fc1_selu<<<(BATCH * DLAT + 255) / 256, 256, 0, stream>>>(z, fc1_w, fc1_b, xb);
  gi0_kernel<<<(BATCH * N3H + 255) / 256, 256, 0, stream>>>(xb, w_ih0, b_ih0, gi0);

  // 4) Recurrence: 120 steps x (3 layers of WMMA GEMM + cell) + fc2/softmax
  dim3 gg(NTILES, 4);
  int  cellg = (BATCH * HDIM + 255) / 256;
  for (int t = 0; t < TSTEPS; ++t) {
    // layer 0 (gi precomputed)
    gemm_step<0><<<gg, 256, 0, stream>>>(h0h, nullptr, wh0, nullptr, b_hh0,
                                         nullptr, ghs, nullptr);
    gru_cell<<<cellg, 256, 0, stream>>>(gi0, N3H, ghs, h0f, h0h);
    // layer 1 (dual GEMM: gi from h0, gh from h1)
    gemm_step<1><<<gg, 256, 0, stream>>>(h1h, h0h, wh1, wi1, b_hh1, b_ih1,
                                         ghs, gis);
    gru_cell<<<cellg, 256, 0, stream>>>(gis, NPAD, ghs, h1f, h1h);
    // layer 2
    gemm_step<1><<<gg, 256, 0, stream>>>(h2h, h1h, wh2, wi2, b_hh2, b_ih2,
                                         ghs, gis);
    gru_cell<<<cellg, 256, 0, stream>>>(gis, NPAD, ghs, h2f, h2h);
    // output head
    fc2_softmax<<<BATCH / 8, 256, 0, stream>>>(h2f, fc2_w, fc2_b, out, t);
  }
}
